// LocalEmbeddingBlock_7894149890566
// MI455X (gfx1250) — matrix operand order repaired and checked by reading, not compiled, not run on hardware
//
#include <hip/hip_runtime.h>
#include <hip/hip_bf16.h>
#include <math.h>

// ---------------------------------------------------------------------------
// Problem constants (from reference)
// ---------------------------------------------------------------------------
#define BB   64
#define NN   512
#define CC   64
#define HIDD 128
#define EE   64
#define KNN  10
#define NHH  4
#define HDD  16
#define BN   (BB * NN)          // 32768 groups
#define BNK  (BN * KNN)         // 327680 edge tokens
#define MTILES (BNK / 16)       // 20480 row tiles

typedef __attribute__((ext_vector_type(16))) _Float16 v16h;
typedef __attribute__((ext_vector_type(8)))  float    v8f;

union FragU { v16h v; unsigned int u[8]; };

// ---------------------------------------------------------------------------
// WMMA fragment loaders (CDNA5 wave32 layouts, ISA 7.12.2)
// A: 16x32 f16.  lane M = lane&15, half = lane>>4.
//   vgpr v<4 : K = 2v + 8*half (+0/1) ; vgpr v>=4 : K = 16 + 2(v-4) + 8*half
// B: 32x16 f16 (B[k][n] = W[n][k]).  lane N = lane&15, half = lane>>4.
//   same K order as A: vgpr v<4 : K = 2v + 8*half ; v>=4 : K = 16 + 2(v-4) + 8*half
// ---------------------------------------------------------------------------
__device__ inline v16h load_a_frag(const _Float16* __restrict__ A, int lda,
                                   size_t m0, int k0, int lane) {
  const int hi = lane >> 4;
  const _Float16* row = A + (m0 + (size_t)(lane & 15)) * (size_t)lda;
  FragU f;
#pragma unroll
  for (int i = 0; i < 8; ++i) {
    int kk = k0 + ((i < 4) ? (2 * i) : (16 + 2 * (i - 4))) + 8 * hi;
    f.u[i] = *reinterpret_cast<const unsigned int*>(row + kk);
  }
  return f.v;
}

__device__ inline v16h load_b_frag(const _Float16* __restrict__ W, int ldw,
                                   int n0, int k0, int lane) {
  const int hi = lane >> 4;
  const _Float16* row = W + (size_t)(n0 + (lane & 15)) * (size_t)ldw + k0 + 8 * hi;
  FragU f;
#pragma unroll
  for (int i = 0; i < 8; ++i)
    f.u[i] = *reinterpret_cast<const unsigned int*>(row + ((i < 4) ? (2 * i) : (16 + 2 * (i - 4))));
  return f.v;
}

__device__ inline float gelu_exact(float x) {
  return 0.5f * x * (1.0f + erff(x * 0.70710678118654752f));
}

// ---------------------------------------------------------------------------
// Fused GEMM stage:  out = epilogue( A[16-rows] @ W^T + bias )
// One workgroup = 16 rows x NCOL cols;  NCOL/16 waves, one 16x16 WMMA tile per
// wave, K-loop in steps of 32.  Epilogue (optional): GELU, *mnb, +residual,
// LayerNorm across NCOL (parallel partial reduction); writes f32 (pre-LN)
// and/or f16 (post-LN).
// ---------------------------------------------------------------------------
template <int KDIM, int NCOL, bool GELU, bool MUL_MNB, bool ADD_RES,
          bool HAS_LN, bool OUT32, bool OUT16>
__global__ __launch_bounds__(NCOL * 2)
void gemm_stage(const _Float16* __restrict__ A, const _Float16* __restrict__ W,
                const float* __restrict__ bias, const float* __restrict__ mnb,
                const float* __restrict__ lng, const float* __restrict__ lnb,
                const float* __restrict__ resin, float* __restrict__ out32,
                _Float16* __restrict__ out16) {
  constexpr int TPR = NCOL / 8;          // threads per row for LN reduction
  __shared__ float tile[16][NCOL];
  __shared__ float psum[16][TPR];
  __shared__ float psq[16][TPR];
  __shared__ float rmean[16];
  __shared__ float rinv[16];

  const int tid  = threadIdx.x;
  const int wave = tid >> 5;
  const int lane = tid & 31;
  const size_t row0 = (size_t)blockIdx.x * 16;
  const int n0 = wave * 16;

  v8f acc = {0.f, 0.f, 0.f, 0.f, 0.f, 0.f, 0.f, 0.f};
#pragma unroll
  for (int k0 = 0; k0 < KDIM; k0 += 32) {
    v16h a = load_a_frag(A, KDIM, row0, k0, lane);
    v16h b = load_b_frag(W, KDIM, n0, k0, lane);
    acc = __builtin_amdgcn_wmma_f32_16x16x32_f16(false, a, false, b,
                                                 (short)0, acc, false, false);
  }

  const int colL = n0 + (lane & 15);
  const float bs = bias[colL];
  const int rhalf = 8 * (lane >> 4);
#pragma unroll
  for (int r = 0; r < 8; ++r) {
    const int rt = r + rhalf;           // row within tile (C/D layout)
    float v = acc[r] + bs;
    if (GELU)    v = gelu_exact(v);
    if (MUL_MNB) v *= mnb[row0 + rt];
    if (ADD_RES) v += resin[(row0 + rt) * NCOL + colL];
    tile[rt][colL] = v;
  }
  __syncthreads();

  if (HAS_LN) {
    // parallel per-row partial sums: 16 rows x TPR threads
    const int rown = tid / TPR;
    const int sub  = tid - rown * TPR;
    float s = 0.f, s2 = 0.f;
#pragma unroll
    for (int c = sub; c < NCOL; c += TPR) {
      const float v = tile[rown][c];
      s  += v;
      s2 += v * v;
    }
    psum[rown][sub] = s;
    psq[rown][sub]  = s2;
    __syncthreads();
    if (tid < 16) {
      float S = 0.f, S2 = 0.f;
#pragma unroll
      for (int t = 0; t < TPR; ++t) { S += psum[tid][t]; S2 += psq[tid][t]; }
      const float m = S * (1.0f / NCOL);
      const float var = fmaxf(S2 * (1.0f / NCOL) - m * m, 0.f);
      rmean[tid] = m;
      rinv[tid]  = rsqrtf(var + 1e-5f);
    }
    __syncthreads();
  }

  for (int i = tid; i < 16 * NCOL; i += NCOL * 2) {
    const int rt = i / NCOL;
    const int c  = i - rt * NCOL;
    float v = tile[rt][c];
    if (OUT32) out32[(row0 + rt) * NCOL + c] = v;
    if (HAS_LN) v = (v - rmean[rt]) * rinv[rt] * lng[c] + lnb[c];
    if (OUT16) out16[(row0 + rt) * NCOL + c] = (_Float16)v;
  }
}

// ---------------------------------------------------------------------------
// k-NN: one block per batch, one thread per query.  Register-resident top-11
// via fully unrolled bubble insertion (strict < keeps stable tie order,
// matching jax top_k index-ascending tie-break).
// ---------------------------------------------------------------------------
__global__ __launch_bounds__(512)
void knn_kernel(const float* __restrict__ pts, int* __restrict__ idxo) {
  __shared__ float sx[NN];
  __shared__ float sy[NN];
  const int b = blockIdx.x;
  for (int j = threadIdx.x; j < NN; j += 512) {
    sx[j] = pts[((size_t)b * NN + j) * 2 + 0];
    sy[j] = pts[((size_t)b * NN + j) * 2 + 1];
  }
  __syncthreads();
  const int q = threadIdx.x;
  const float px = sx[q], py = sy[q];
  float bd[KNN + 1];
  int   bi[KNN + 1];
#pragma unroll
  for (int t = 0; t <= KNN; ++t) { bd[t] = 3.4e38f; bi[t] = -1; }
  for (int j = 0; j < NN; ++j) {
    const float dx = px - sx[j], dy = py - sy[j];
    const float d = dx * dx + dy * dy;
    if (d < bd[KNN]) {
      bd[KNN] = d;
      bi[KNN] = j;
#pragma unroll
      for (int t = KNN; t >= 1; --t) {
        const bool sw = bd[t] < bd[t - 1];      // strict: stable on ties
        const float td = sw ? bd[t] : bd[t - 1];
        bd[t]     = sw ? bd[t - 1] : bd[t];
        bd[t - 1] = td;
        const int ti = sw ? bi[t] : bi[t - 1];
        bi[t]     = sw ? bi[t - 1] : bi[t];
        bi[t - 1] = ti;
      }
    }
  }
#pragma unroll
  for (int k = 0; k < KNN; ++k)
    idxo[((size_t)b * NN + q) * KNN + k] = bi[k + 1];  // drop self
}

// ---------------------------------------------------------------------------
// Edge features: builds loc = [ (ctr-nb)*mnb (64), logm, logdr, logkt ] padded
// to 96 f16 columns; also writes mnb per edge row.
// ---------------------------------------------------------------------------
__global__ __launch_bounds__(64)
void edge_kernel(const float* __restrict__ feats, const float* __restrict__ mask,
                 const int* __restrict__ idx, float* __restrict__ mnbo,
                 _Float16* __restrict__ Aloc) {
  const int g = blockIdx.x;           // b*N + n
  const int c = threadIdx.x;          // channel 0..63
  const int b = g >> 9;               // N == 512
  const float* ctr = feats + (size_t)g * CC;
  const float cc = ctr[c];
  const float mc = mask[g];
  for (int k = 0; k < KNN; ++k) {
    const int j = idx[(size_t)g * KNN + k];
    const size_t gj = (size_t)b * NN + j;
    const float mn = mask[gj] * mc;
    const float* nbp = feats + gj * CC;
    const size_t row = (size_t)g * KNN + k;
    _Float16* ar = Aloc + row * 96;
    ar[c] = (_Float16)((cc - nbp[c]) * mn);
    if (c >= 3 && c < 32) ar[64 + c] = (_Float16)0.f;  // pad cols 67..95
    if (c == 0) {
      const float deta = ctr[0] - nbp[0];
      const float dphi = ctr[1] - nbp[1];
      const float dr2 = deta * deta + dphi * dphi;
      const float logdr = 0.5f * logf(dr2 > 0.f ? dr2 : 1.f) * mn;
      const float dr = sqrtf(dr2) * mn;
      const float ec = expf(ctr[2]), en = expf(nbp[2]);
      const float kt = fminf(ec, en) * dr;
      const float logkt = logf(kt > 0.f ? kt : 1.f) * mn;
      const float m2 = 2.f * ec * en * (coshf(deta) - cosf(dphi));
      const float logm = logf(m2 > 0.f ? m2 : 1.f) * mn;
      ar[64] = (_Float16)logm;
      ar[65] = (_Float16)logdr;
      ar[66] = (_Float16)logkt;
      mnbo[row] = mn;
    }
  }
}

// ---------------------------------------------------------------------------
// Per-group attention: 4 heads x 10 queries over 10 keys (VALU; tiny).
// ---------------------------------------------------------------------------
__global__ __launch_bounds__(64)
void attn_kernel(const _Float16* __restrict__ qkv, _Float16* __restrict__ o16) {
  const int g = blockIdx.x;
  const int t = threadIdx.x;
  if (t >= NHH * KNN) return;
  const int h = t / KNN, q = t - h * KNN;
  const _Float16* base = qkv + (size_t)g * KNN * (3 * EE);
  float qv[HDD];
#pragma unroll
  for (int d = 0; d < HDD; ++d) qv[d] = (float)base[q * 192 + h * HDD + d];
  float s[KNN];
  float mx = -3.4e38f;
  for (int kk = 0; kk < KNN; ++kk) {
    const _Float16* kp = base + kk * 192 + EE + h * HDD;
    float a = 0.f;
#pragma unroll
    for (int d = 0; d < HDD; ++d) a += qv[d] * (float)kp[d];
    s[kk] = a * 0.25f;  // 1/sqrt(16)
    mx = fmaxf(mx, s[kk]);
  }
  float sum = 0.f;
  for (int kk = 0; kk < KNN; ++kk) { s[kk] = expf(s[kk] - mx); sum += s[kk]; }
  const float inv = 1.f / sum;
  float o[HDD];
#pragma unroll
  for (int d = 0; d < HDD; ++d) o[d] = 0.f;
  for (int kk = 0; kk < KNN; ++kk) {
    const _Float16* vp = base + kk * 192 + 2 * EE + h * HDD;
    const float a = s[kk] * inv;
#pragma unroll
    for (int d = 0; d < HDD; ++d) o[d] += a * (float)vp[d];
  }
  _Float16* op = o16 + ((size_t)g * KNN + q) * EE + h * HDD;
#pragma unroll
  for (int d = 0; d < HDD; ++d) op[d] = (_Float16)o[d];
}

// ---------------------------------------------------------------------------
// Masked mean over K neighbors + final LayerNorm (n3) -> d_out (B,N,E) f32.
// ---------------------------------------------------------------------------
__global__ __launch_bounds__(64)
void reduce_kernel(const float* __restrict__ att, const float* __restrict__ mnb,
                   const float* __restrict__ mask, const float* __restrict__ g3,
                   const float* __restrict__ b3, float* __restrict__ out) {
  __shared__ float sv[EE];
  __shared__ float sm, si;
  const int g = blockIdx.x;
  const int c = threadIdx.x;
  float s = 0.f;
  for (int k = 0; k < KNN; ++k) s += att[((size_t)g * KNN + k) * EE + c];
  float den = (float)KNN * 1e-9f;
  for (int k = 0; k < KNN; ++k) den += mnb[(size_t)g * KNN + k];
  const float val = s / den * mask[g];
  sv[c] = val;
  __syncthreads();
  if (c == 0) {
    float m = 0.f;
    for (int i = 0; i < EE; ++i) m += sv[i];
    m *= (1.0f / EE);
    float v = 0.f;
    for (int i = 0; i < EE; ++i) { float d = sv[i] - m; v += d * d; }
    v *= (1.0f / EE);
    sm = m;
    si = rsqrtf(v + 1e-5f);
  }
  __syncthreads();
  out[(size_t)g * EE + c] = (sv[c] - sm) * si * g3[c] + b3[c];
}

// ---------------------------------------------------------------------------
// Weight conversion helpers
// ---------------------------------------------------------------------------
__global__ void f32_to_f16(const float* __restrict__ s, _Float16* __restrict__ d,
                           int n) {
  const int i = blockIdx.x * blockDim.x + threadIdx.x;
  if (i < n) d[i] = (_Float16)s[i];
}

__global__ void pack_w1(const float* __restrict__ w, _Float16* __restrict__ d) {
  const int i = blockIdx.x * blockDim.x + threadIdx.x;  // 128*96
  if (i < HIDD * 96) {
    const int r = i / 96, c = i - r * 96;
    d[i] = (c < 67) ? (_Float16)w[r * 67 + c] : (_Float16)0.f;
  }
}

// ---------------------------------------------------------------------------
extern "C" void kernel_launch(void* const* d_in, const int* in_sizes, int n_in,
                              void* d_out, int out_size, void* d_ws,
                              size_t ws_size, hipStream_t stream) {
  (void)in_sizes; (void)n_in; (void)out_size; (void)ws_size;
  const float* points   = (const float*)d_in[0];
  const float* features = (const float*)d_in[1];
  const float* mask     = (const float*)d_in[2];
  const float* m1_fw1 = (const float*)d_in[3];
  const float* m1_fb1 = (const float*)d_in[4];
  const float* m1_lg  = (const float*)d_in[5];
  const float* m1_lb  = (const float*)d_in[6];
  const float* m1_fw2 = (const float*)d_in[7];
  const float* m1_fb2 = (const float*)d_in[8];
  const float* n1g = (const float*)d_in[9];
  const float* n1b = (const float*)d_in[10];
  const float* n2g = (const float*)d_in[11];
  const float* n2b = (const float*)d_in[12];
  const float* n3g = (const float*)d_in[13];
  const float* n3b = (const float*)d_in[14];
  const float* aw = (const float*)d_in[15];
  const float* ab = (const float*)d_in[16];
  const float* ow = (const float*)d_in[17];
  const float* ob = (const float*)d_in[18];
  const float* m2_fw1 = (const float*)d_in[19];
  const float* m2_fb1 = (const float*)d_in[20];
  const float* m2_lg  = (const float*)d_in[21];
  const float* m2_lb  = (const float*)d_in[22];
  const float* m2_fw2 = (const float*)d_in[23];
  const float* m2_fb2 = (const float*)d_in[24];

  // workspace carve-up
  char* base = (char*)d_ws;
  size_t off = 0;
  auto carve = [&](size_t bytes) -> char* {
    char* p = base + off;
    off = (off + bytes + 255) & ~(size_t)255;
    return p;
  };
  int*       idxb  = (int*)carve((size_t)BNK * 4);
  float*     mnb   = (float*)carve((size_t)BNK * 4);
  _Float16*  Aloc  = (_Float16*)carve((size_t)BNK * 96 * 2);
  _Float16*  H16   = (_Float16*)carve((size_t)BNK * HIDD * 2);
  _Float16*  XN16  = (_Float16*)carve((size_t)BNK * EE * 2);
  _Float16*  QKV16 = (_Float16*)carve((size_t)BNK * 192 * 2);
  _Float16*  O16   = (_Float16*)carve((size_t)BNK * EE * 2);
  float*     ATT   = (float*)carve((size_t)BNK * EE * 4);
  _Float16*  AN16  = (_Float16*)carve((size_t)BNK * EE * 2);
  _Float16*  W1p   = (_Float16*)carve((size_t)HIDD * 96 * 2);
  _Float16*  W2h   = (_Float16*)carve((size_t)EE * HIDD * 2);
  _Float16*  AWh   = (_Float16*)carve((size_t)192 * EE * 2);
  _Float16*  OWh   = (_Float16*)carve((size_t)EE * EE * 2);
  _Float16*  M2W1h = (_Float16*)carve((size_t)HIDD * EE * 2);
  _Float16*  M2W2h = (_Float16*)carve((size_t)EE * HIDD * 2);

  // weight conversions
  pack_w1<<<(HIDD * 96 + 255) / 256, 256, 0, stream>>>(m1_fw1, W1p);
  f32_to_f16<<<(EE * HIDD + 255) / 256, 256, 0, stream>>>(m1_fw2, W2h, EE * HIDD);
  f32_to_f16<<<(192 * EE + 255) / 256, 256, 0, stream>>>(aw, AWh, 192 * EE);
  f32_to_f16<<<(EE * EE + 255) / 256, 256, 0, stream>>>(ow, OWh, EE * EE);
  f32_to_f16<<<(HIDD * EE + 255) / 256, 256, 0, stream>>>(m2_fw1, M2W1h, HIDD * EE);
  f32_to_f16<<<(EE * HIDD + 255) / 256, 256, 0, stream>>>(m2_fw2, M2W2h, EE * HIDD);

  // 1) kNN + edge features
  knn_kernel<<<BB, 512, 0, stream>>>(points, idxb);
  edge_kernel<<<BN, 64, 0, stream>>>(features, mask, idxb, mnb, Aloc);

  // 2) MLP1: h = LN(gelu(loc @ fw1^T + fb1))            -> H16 [BNK,128]
  gemm_stage<96, 128, true, false, false, true, false, true>
      <<<MTILES, 256, 0, stream>>>(Aloc, W1p, m1_fb1, nullptr, m1_lg, m1_lb,
                                   nullptr, nullptr, H16);
  // 3) x = (h @ fw2^T + fb2)*mnb ; xn = LN_n1(x)        -> XN16 [BNK,64]
  gemm_stage<128, 64, false, true, false, true, false, true>
      <<<MTILES, 128, 0, stream>>>(H16, W2h, m1_fb2, mnb, n1g, n1b,
                                   nullptr, nullptr, XN16);
  // 4) qkv = xn @ aw^T + ab                              -> QKV16 [BNK,192]
  gemm_stage<64, 192, false, false, false, false, false, true>
      <<<MTILES, 384, 0, stream>>>(XN16, AWh, ab, nullptr, nullptr, nullptr,
                                   nullptr, nullptr, QKV16);
  // 5) per-group softmax attention                       -> O16 [BNK,64]
  attn_kernel<<<BN, 64, 0, stream>>>(QKV16, O16);
  // 6) att = (o @ ow^T + ob)*mnb ; an = LN_n2(att)       -> ATT f32, AN16 f16
  gemm_stage<64, 64, false, true, false, true, true, true>
      <<<MTILES, 128, 0, stream>>>(O16, OWh, ob, mnb, n2g, n2b,
                                   nullptr, ATT, AN16);
  // 7) h2 = LN(gelu(an @ m2_fw1^T + m2_fb1))             -> H16 (reuse)
  gemm_stage<64, 128, true, false, false, true, false, true>
      <<<MTILES, 256, 0, stream>>>(AN16, M2W1h, m2_fb1, nullptr, m2_lg, m2_lb,
                                   nullptr, nullptr, H16);
  // 8) att += (h2 @ m2_fw2^T + m2_fb2)*mnb               -> ATT (in place)
  gemm_stage<128, 64, false, true, true, false, true, false>
      <<<MTILES, 128, 0, stream>>>(H16, M2W2h, m2_fb2, mnb, nullptr, nullptr,
                                   ATT, ATT, nullptr);
  // 9) masked mean over K + LN_n3                        -> d_out (B,N,64) f32
  reduce_kernel<<<BN, 64, 0, stream>>>(ATT, mnb, mask, n3g, n3b, (float*)d_out);
}
